// DeepLinearAttention_91079076479301
// MI455X (gfx1250) — compile-verified, matrix operand order checked by reading, not run on hardware
//
#include <hip/hip_runtime.h>
#include <hip/hip_bf16.h>

// ---------------------------------------------------------------------------
// DeepLinearAttention for MI455X (gfx1250), chunked linear-attention formulation.
// v_wmma_f32_16x16x32_bf16 everywhere; fp32 state in LDS; chunk staging via the
// Tensor Data Mover (TENSORcnt) when available, else async global->LDS
// (ASYNCcnt); double buffering on the sequential recurrence.
// ---------------------------------------------------------------------------

typedef __attribute__((ext_vector_type(16))) __bf16 v16bf;
typedef __attribute__((ext_vector_type(8)))  float  v8f;

#define WMMA_BF16(a, b, c) \
    __builtin_amdgcn_wmma_f32_16x16x32_bf16(false, (a), false, (b), (short)0, (c), false, false)

static constexpr int Db  = 128;   // model dim
static constexpr int CCH = 32;    // recurrence chunk (tokens)

// --------------------------- async global->LDS -----------------------------
#if defined(__AMDGCN__) && \
    __has_builtin(__builtin_amdgcn_global_load_async_to_lds_b128) && \
    __has_builtin(__builtin_amdgcn_s_wait_asynccnt)
#define DLA_ASYNC 1
#else
#define DLA_ASYNC 0
#endif

#if DLA_ASYNC
typedef int dla_v4i __attribute__((vector_size(16)));
typedef __attribute__((address_space(1))) dla_v4i* dla_gptr;
typedef __attribute__((address_space(3))) dla_v4i* dla_lptr;
#endif

__device__ __forceinline__ void dla_copy_b128(const uint4* g, uint4* l) {
#if DLA_ASYNC
    __builtin_amdgcn_global_load_async_to_lds_b128((dla_gptr)g, (dla_lptr)l, 0, 0);
#else
    *l = *g;
#endif
}

__device__ __forceinline__ void dla_async_wait() {
#if DLA_ASYNC
    __builtin_amdgcn_s_wait_asynccnt(0);
#endif
}

// --------------------------- Tensor Data Mover -----------------------------
#if defined(__AMDGCN__) && \
    __has_builtin(__builtin_amdgcn_tensor_load_to_lds) && \
    __has_builtin(__builtin_amdgcn_s_wait_tensorcnt)
#define DLA_TDM 1
#else
#define DLA_TDM 0
#endif

#if DLA_TDM
typedef unsigned int dla_u32x4 __attribute__((vector_size(16)));
typedef int          dla_i32x4 __attribute__((vector_size(16)));
typedef int          dla_i32x8 __attribute__((vector_size(32)));

// One TDM descriptor moves a 32x128 bf16 tile (row-major, stride 128) into LDS.
__device__ __forceinline__ void dla_tdm_load_tile(const __bf16* gsrc, __bf16* ldst) {
    unsigned long long ga = (unsigned long long)gsrc;
    unsigned lds = (unsigned)(unsigned long long)
        ( __attribute__((address_space(3))) __bf16*)ldst;

    dla_u32x4 g0;
    g0[0] = 1u;                                           // count=1, user D#
    g0[1] = lds;                                          // lds_addr
    g0[2] = (unsigned)ga;                                 // global_addr[31:0]
    g0[3] = (unsigned)((ga >> 32) & 0x1FFFFFFu) | (2u << 30);  // addr[56:32] | type=2

    dla_i32x8 g1;
    g1[0] = 1 << 16;                 // data_size=1 (2 bytes/elem)
    g1[1] = (int)(128u << 16);       // tensor_dim0 = 128 (bits 79:48 low part)
    g1[2] = (int)(CCH << 16);        // tensor_dim1 = 32  (bits 111:80 low part)
    g1[3] = (int)(128u << 16);       // tile_dim0 = 128   (bits 127:112)
    g1[4] = CCH;                     // tile_dim1 = 32, tile_dim2 = 0
    g1[5] = 128;                     // tensor_dim0_stride = 128 (elems)
    g1[6] = 0;
    g1[7] = 0;

    dla_i32x4 z4 = {0, 0, 0, 0};
#if __clang_major__ >= 23
    dla_i32x8 z8 = {0, 0, 0, 0, 0, 0, 0, 0};
    __builtin_amdgcn_tensor_load_to_lds(g0, g1, z4, z4, z8, 0);
#else
    __builtin_amdgcn_tensor_load_to_lds(g0, g1, z4, z4, 0);
#endif
}
#endif

// --------------------------- clamp to +-1e20 -------------------------------
__device__ __forceinline__ float dla_clamp(float v) {
    return __builtin_amdgcn_fmed3f(v, -1e20f, 1e20f);     // single v_med3
}

// ---------------------------------------------------------------------------
// fp32 -> bf16 elementwise (weights)
// ---------------------------------------------------------------------------
__global__ __launch_bounds__(256) void dla_f32_to_bf16(const float* __restrict__ s,
                                                       __bf16* __restrict__ d, int n) {
    int i = blockIdx.x * 256 + threadIdx.x;
    if (i < n) d[i] = (__bf16)s[i];
}

// ---------------------------------------------------------------------------
// LayerNorm (eps=1e-5), one row (D=128) per 128-thread block, bf16 output
// ---------------------------------------------------------------------------
__global__ __launch_bounds__(128) void dla_layernorm(const float* __restrict__ x,
                                                     const float* __restrict__ g,
                                                     const float* __restrict__ beta,
                                                     __bf16* __restrict__ xn) {
    const int row = blockIdx.x;
    const int t   = threadIdx.x;
    __shared__ float red[128];

    float v = x[(size_t)row * Db + t];
    red[t] = v;
    __syncthreads();
    #pragma unroll
    for (int s = 64; s > 0; s >>= 1) {
        if (t < s) red[t] += red[t + s];
        __syncthreads();
    }
    float mu = red[0] * (1.0f / 128.0f);
    __syncthreads();
    float dlt = v - mu;
    red[t] = dlt * dlt;
    __syncthreads();
    #pragma unroll
    for (int s = 64; s > 0; s >>= 1) {
        if (t < s) red[t] += red[t + s];
        __syncthreads();
    }
    float var = red[0] * (1.0f / 128.0f);
    float out = dlt * rsqrtf(var + 1e-5f) * g[t] + beta[t];
    xn[(size_t)row * Db + t] = (__bf16)out;
}

// ---------------------------------------------------------------------------
// WMMA GEMM:  C[M,128] = act(A[M,128] @ W[128,128]^T + bias) (+ addend)
// A bf16, W bf16 row-major [N=128, K=128].  ACT: 0 none, 1 elu+1, 2 relu.
// Compile-time specialization -> straight-line epilogue.
// ---------------------------------------------------------------------------
template <int ACT, bool BIAS, bool ADD, bool OUTB, bool OUTF>
__global__ __launch_bounds__(256) void dla_gemm_nt(const __bf16* __restrict__ A,
                                                   const __bf16* __restrict__ W,
                                                   const float* __restrict__ bias,
                                                   const float* __restrict__ addend,
                                                   __bf16* __restrict__ outb,
                                                   float* __restrict__ outf) {
    __shared__ __attribute__((aligned(32))) __bf16 As[32 * Db];

    const int m0   = blockIdx.x * 32;
    const int tid  = threadIdx.x;
    const int lane = tid & 31;
    const int w    = tid >> 5;
    const int lrow = lane & 15;
    const int lhi  = lane >> 4;

    // Stage 32x128 bf16 A tile into LDS (8 KB) via async b128 copies.
    const uint4* Ag  = (const uint4*)(A + (size_t)m0 * Db);
    uint4*       AsV = (uint4*)As;
    dla_copy_b128(Ag + tid,       AsV + tid);
    dla_copy_b128(Ag + tid + 256, AsV + tid + 256);
    dla_async_wait();
    __syncthreads();

    #pragma unroll
    for (int tt = 0; tt < 2; ++tt) {
        const int tile = w * 2 + tt;
        const int mi   = tile >> 3;     // 0..1
        const int ni   = tile & 7;      // 0..7
        const __bf16* Ap = As + (mi * 16 + lrow) * Db + lhi * 16;
        const __bf16* Wp = W + (ni * 16 + lrow) * Db + lhi * 16;
        __builtin_prefetch(Wp, 0, 3);   // L2-hot weights -> global_prefetch_b8

        v8f acc = {};
        #pragma unroll
        for (int k0 = 0; k0 < Db; k0 += 32) {
            v16bf a = *(const v16bf*)(Ap + k0);
            v16bf b = *(const v16bf*)(Wp + k0);
            acc = WMMA_BF16(a, b, acc);
        }
        #pragma unroll
        for (int r = 0; r < 8; ++r) {
            const int m = m0 + mi * 16 + r + 8 * lhi;
            const int n = ni * 16 + lrow;
            float v = acc[r];
            if (BIAS) v += bias[n];
            if (ACT == 1) v = (v > 0.0f) ? v + 1.0f : __expf(v);   // elu(v)+1
            if (ACT == 2) v = (v > 0.0f) ? v : 0.0f;               // relu
            if (ADD) v += addend[(size_t)m * Db + n];
            if (OUTF) outf[(size_t)m * Db + n] = v;
            if (OUTB) outb[(size_t)m * Db + n] = (__bf16)v;
        }
    }
}

// ---------------------------------------------------------------------------
// Chunked linear-attention recurrence. One block per batch element.
// Chunk tiles staged by the TDM (TENSORcnt) or per-lane async loads (ASYNCcnt),
// double buffered so staging overlaps the sequential state-update chain.
// ---------------------------------------------------------------------------
__global__ __launch_bounds__(256) void dla_linattn(const __bf16* __restrict__ Kg,
                                                   const __bf16* __restrict__ Qg,
                                                   const __bf16* __restrict__ Vg,
                                                   const float* __restrict__ S0,
                                                   const float* __restrict__ Z0,
                                                   __bf16* __restrict__ outb,
                                                   int T) {
    const int b    = blockIdx.x;
    const int tid  = threadIdx.x;
    const int lane = tid & 31;
    const int w    = tid >> 5;
    const int lrow = lane & 15;
    const int lhi  = lane >> 4;

    // Sst[j*128 + i] = S[i][j]  (column-major so Q@S B-fragments are contiguous)
    __shared__ __attribute__((aligned(32))) float  Sst[Db * Db];        // 64 KB
    __shared__ __attribute__((aligned(32))) __bf16 Kc[2][CCH * Db];     // 2 x 8 KB
    __shared__ __attribute__((aligned(32))) __bf16 Qc[2][CCH * Db];     // 2 x 8 KB
    __shared__ __attribute__((aligned(32))) __bf16 Vc[2][CCH * Db];     // 2 x 8 KB
    __shared__ __attribute__((aligned(32))) __bf16 KcT[Db * CCH];       // 8 KB, [d][t]
    __shared__ __attribute__((aligned(32))) __bf16 VcT[Db * CCH];       // 8 KB, [d][t]
    __shared__ __attribute__((aligned(32))) __bf16 Ab[CCH * CCH];       // 2 KB
    __shared__ float Zf[Db];
    __shared__ float den[CCH];

    for (int i = tid; i < Db * Db; i += 256)
        Sst[i] = S0[(size_t)b * Db * Db + (size_t)(i & 127) * Db + (i >> 7)];
    if (tid < Db) Zf[tid] = Z0[(size_t)b * Db + tid];

    const size_t base = (size_t)b * T * Db;

    auto issue_chunk = [&](int c0, int buf) {
#if DLA_TDM
        if (w == 0) {   // one wave issues three tile descriptors (TENSORcnt)
            dla_tdm_load_tile(Kg + base + (size_t)c0 * Db, Kc[buf]);
            dla_tdm_load_tile(Qg + base + (size_t)c0 * Db, Qc[buf]);
            dla_tdm_load_tile(Vg + base + (size_t)c0 * Db, Vc[buf]);
        }
#else
        const uint4* kg = (const uint4*)(Kg + base + (size_t)c0 * Db);
        const uint4* qg = (const uint4*)(Qg + base + (size_t)c0 * Db);
        const uint4* vg = (const uint4*)(Vg + base + (size_t)c0 * Db);
        uint4* kd = (uint4*)Kc[buf];
        uint4* qd = (uint4*)Qc[buf];
        uint4* vd = (uint4*)Vc[buf];
        #pragma unroll
        for (int i = 0; i < 2; ++i) {
            dla_copy_b128(kg + tid + i * 256, kd + tid + i * 256);
            dla_copy_b128(qg + tid + i * 256, qd + tid + i * 256);
            dla_copy_b128(vg + tid + i * 256, vd + tid + i * 256);
        }
#endif
    };

    issue_chunk(0, 0);      // prime the pipeline
    int cb = 0;

    for (int c0 = 0; c0 < T; c0 += CCH, cb ^= 1) {
#if DLA_TDM
        if (w == 0) __builtin_amdgcn_s_wait_tensorcnt(0);
#else
        dla_async_wait();
#endif
        __syncthreads();    // staged chunk visible to all waves

        if (c0 + CCH < T) issue_chunk(c0 + CCH, cb ^ 1);   // prefetch next chunk

        const __bf16* Kcb = Kc[cb];
        const __bf16* Qcb = Qc[cb];
        const __bf16* Vcb = Vc[cb];

        if (w < 4) {
            // ---- attn = tril(Qc Kc^T), bf16 to LDS (waves 0-3) ----
            const int mi = w >> 1, ni = w & 1;
            const __bf16* Qp = Qcb + (mi * 16 + lrow) * Db + lhi * 16;
            const __bf16* Kp = Kcb + (ni * 16 + lrow) * Db + lhi * 16;
            v8f acc = {};
            #pragma unroll
            for (int k0 = 0; k0 < Db; k0 += 32) {
                v16bf a  = *(const v16bf*)(Qp + k0);
                v16bf bb = *(const v16bf*)(Kp + k0);
                acc = WMMA_BF16(a, bb, acc);
            }
            #pragma unroll
            for (int r = 0; r < 8; ++r) {
                const int m = mi * 16 + r + 8 * lhi;
                const int n = ni * 16 + lrow;
                float v = (n <= m) ? acc[r] : 0.0f;     // inclusive causal mask
                Ab[m * CCH + n] = (__bf16)v;
            }
        } else {
            // ---- transpose K,V into [d][t] layout (waves 4-7, one column each) ----
            const int d = tid - 128;                    // 0..127
            v16bf k0v, k1v, v0v, v1v;
            #pragma unroll
            for (int t = 0; t < 16; ++t) {
                k0v[t] = Kcb[t * Db + d];
                k1v[t] = Kcb[(16 + t) * Db + d];
                v0v[t] = Vcb[t * Db + d];
                v1v[t] = Vcb[(16 + t) * Db + d];
            }
            *(v16bf*)(KcT + d * CCH)      = k0v;
            *(v16bf*)(KcT + d * CCH + 16) = k1v;
            *(v16bf*)(VcT + d * CCH)      = v0v;
            *(v16bf*)(VcT + d * CCH + 16) = v1v;
        }
        __syncthreads();

        // ---- den (wave 7) + num accumulation (all waves) ----
        if (w == 7) {
            const int t = lane;                          // 32 tokens, wave32
            float s = 1e-5f;                             // EPS_DEN
            for (int j = 0; j < Db; ++j) s += (float)Qcb[t * Db + j] * Zf[j];
            for (int s2 = 0; s2 < CCH; ++s2) s += (float)Ab[t * CCH + s2];
            den[t] = s;
        }

        v8f num[2];
        #pragma unroll
        for (int tt = 0; tt < 2; ++tt) {
            const int tile = w * 2 + tt;
            const int mi = tile >> 3, ni = tile & 7;
            const __bf16* Qp = Qcb + (mi * 16 + lrow) * Db + lhi * 16;
            v8f acc = {};
            // inter-chunk: Qc @ S_prev   (B fragment built from fp32 state)
            #pragma unroll
            for (int k0 = 0; k0 < Db; k0 += 32) {
                v16bf a = *(const v16bf*)(Qp + k0);
                const float* sp = Sst + (ni * 16 + lrow) * Db + k0 + lhi * 16;
                v16bf bb;
                #pragma unroll
                for (int kk = 0; kk < 16; ++kk) bb[kk] = (__bf16)sp[kk];
                acc = WMMA_BF16(a, bb, acc);
            }
            // intra-chunk: attn @ Vc  (B fragment = contiguous VcT row)
            {
                v16bf aa = *(const v16bf*)(Ab + (mi * 16 + lrow) * CCH + lhi * 16);
                v16bf bv = *(const v16bf*)(VcT + (ni * 16 + lrow) * CCH + lhi * 16);
                acc = WMMA_BF16(aa, bv, acc);
            }
            num[tt] = acc;
        }
        __syncthreads();   // den ready; all Q@S reads of Sst complete

        // ---- epilogue: out = num/den ----
        #pragma unroll
        for (int tt = 0; tt < 2; ++tt) {
            const int tile = w * 2 + tt;
            const int mi = tile >> 3, ni = tile & 7;
            #pragma unroll
            for (int r = 0; r < 8; ++r) {
                const int ml = mi * 16 + r + 8 * lhi;        // token in chunk
                const int n  = ni * 16 + lrow;
                float v = num[tt][r] / den[ml];
                outb[base + (size_t)(c0 + ml) * Db + n] = (__bf16)v;
            }
        }

        // ---- Z += colsum(K), clamped (contiguous KcT row) ----
        if (tid < Db) {
            float z = Zf[tid];
            #pragma unroll
            for (int t = 0; t < CCH; ++t) z += (float)KcT[tid * CCH + t];
            Zf[tid] = dla_clamp(z);
        }

        // ---- S += Kc^T @ Vc (wave w owns i-rows w*16..w*16+15), clamped ----
        {
            v16bf akt = *(const v16bf*)(KcT + (w * 16 + lrow) * CCH + lhi * 16);
            #pragma unroll
            for (int jb = 0; jb < 8; ++jb) {
                v16bf bv = *(const v16bf*)(VcT + (jb * 16 + lrow) * CCH + lhi * 16);
                v8f cc;
                #pragma unroll
                for (int r = 0; r < 8; ++r)
                    cc[r] = Sst[(jb * 16 + lrow) * Db + w * 16 + r + 8 * lhi];
                cc = WMMA_BF16(akt, bv, cc);
                #pragma unroll
                for (int r = 0; r < 8; ++r)
                    Sst[(jb * 16 + lrow) * Db + w * 16 + r + 8 * lhi] = dla_clamp(cc[r]);
            }
        }
        __syncthreads();   // protect Zf/Sst/KcT/VcT (and buffer rotation)
    }
}

// ---------------------------------------------------------------------------
// Host-side orchestration
// ---------------------------------------------------------------------------
extern "C" void kernel_launch(void* const* d_in, const int* in_sizes, int n_in,
                              void* d_out, int out_size, void* d_ws, size_t ws_size,
                              hipStream_t stream) {
    (void)in_sizes; (void)n_in; (void)out_size; (void)ws_size;
    const int B = 8, T = 512;
    const int M = B * T;                       // 4096 rows
    const size_t nTok = (size_t)M * Db;        // 524288

    const float* z = (const float*)d_in[0];
    const float* S0[2] = { (const float*)d_in[1], (const float*)d_in[3] };
    const float* Z0[2] = { (const float*)d_in[2], (const float*)d_in[4] };
    // per-layer params at 5 + 11*li: g,beta,Wk,Wq,Wv,W1,b1,W2,b2,Ws,bs
    const float* Wu = (const float*)d_in[27];
    const float* bu = (const float*)d_in[28];

    // workspace carve (256B aligned regions)
    char* p = (char*)d_ws;
    auto carve = [&](size_t bytes) -> void* {
        void* r = (void*)p;
        p += (bytes + 255) & ~(size_t)255;
        return r;
    };
    __bf16* xn_bf   = (__bf16*)carve(nTok * 2);
    __bf16* Kb      = (__bf16*)carve(nTok * 2);
    __bf16* Qb      = (__bf16*)carve(nTok * 2);
    __bf16* Vb      = (__bf16*)carve(nTok * 2);
    __bf16* attn_bf = (__bf16*)carve(nTok * 2);
    __bf16* h1_bf   = (__bf16*)carve(nTok * 2);
    __bf16* x_bf    = (__bf16*)carve(nTok * 2);
    float*  h2_f    = (float*) carve(nTok * 4);
    float*  x_f     = (float*) carve(nTok * 4);
    __bf16* wbf     = (__bf16*)carve((size_t)13 * Db * Db * 2);

    // convert all 13 weight matrices to bf16 (Wk,Wq,Wv,W1,W2,Ws per layer; Wu)
    const int wIdx[6] = { 2, 3, 4, 5, 7, 9 };
    for (int li = 0; li < 2; ++li)
        for (int j = 0; j < 6; ++j)
            dla_f32_to_bf16<<<(Db * Db + 255) / 256, 256, 0, stream>>>(
                (const float*)d_in[5 + 11 * li + wIdx[j]],
                wbf + (size_t)(li * 6 + j) * Db * Db, Db * Db);
    dla_f32_to_bf16<<<(Db * Db + 255) / 256, 256, 0, stream>>>(
        Wu, wbf + (size_t)12 * Db * Db, Db * Db);

    const float* xin = z;
    for (int li = 0; li < 2; ++li) {
        const int pb = 5 + 11 * li;
        const float* g    = (const float*)d_in[pb + 0];
        const float* beta = (const float*)d_in[pb + 1];
        const float* b1   = (const float*)d_in[pb + 6];
        const float* b2   = (const float*)d_in[pb + 8];
        const float* bs   = (const float*)d_in[pb + 10];
        __bf16* Wk = wbf + (size_t)(li * 6 + 0) * Db * Db;
        __bf16* Wq = wbf + (size_t)(li * 6 + 1) * Db * Db;
        __bf16* Wv = wbf + (size_t)(li * 6 + 2) * Db * Db;
        __bf16* W1 = wbf + (size_t)(li * 6 + 3) * Db * Db;
        __bf16* W2 = wbf + (size_t)(li * 6 + 4) * Db * Db;
        __bf16* Ws = wbf + (size_t)(li * 6 + 5) * Db * Db;

        dla_layernorm<<<M, 128, 0, stream>>>(xin, g, beta, xn_bf);
        // K,Q: elu+1, no bias, bf16 out
        dla_gemm_nt<1, false, false, true, false><<<M / 32, 256, 0, stream>>>(
            xn_bf, Wk, nullptr, nullptr, Kb, nullptr);
        dla_gemm_nt<1, false, false, true, false><<<M / 32, 256, 0, stream>>>(
            xn_bf, Wq, nullptr, nullptr, Qb, nullptr);
        // V: plain, bf16 out
        dla_gemm_nt<0, false, false, true, false><<<M / 32, 256, 0, stream>>>(
            xn_bf, Wv, nullptr, nullptr, Vb, nullptr);

        dla_linattn<<<B, 256, 0, stream>>>(Kb, Qb, Vb, S0[li], Z0[li], attn_bf, T);

        // FFN: relu(W1) -> bf16, relu(W2) -> f32, shortcut adds h2
        dla_gemm_nt<2, true, false, true, false><<<M / 32, 256, 0, stream>>>(
            attn_bf, W1, b1, nullptr, h1_bf, nullptr);
        dla_gemm_nt<2, true, false, false, true><<<M / 32, 256, 0, stream>>>(
            h1_bf, W2, b2, nullptr, nullptr, h2_f);
        dla_gemm_nt<0, true, true, true, true><<<M / 32, 256, 0, stream>>>(
            xn_bf, Ws, bs, h2_f, x_bf, x_f);
        xin = x_f;
    }
    // final projection -> fp32 output
    dla_gemm_nt<0, true, false, false, true><<<M / 32, 256, 0, stream>>>(
        x_bf, wbf + (size_t)12 * Db * Db, bu, nullptr, nullptr, (float*)d_out);
}
